// Model_84954453115692
// MI455X (gfx1250) — matrix operand order repaired and checked by reading, not compiled, not run on hardware
//
#include <hip/hip_runtime.h>
#include <hip/hip_bf16.h>

#define DI __device__ __forceinline__

typedef __bf16 bf16_t;
typedef __attribute__((ext_vector_type(16))) __bf16 v16bf;
typedef __attribute__((ext_vector_type(8)))  float  v8f;

// ---------------- problem dimensions ----------------
constexpr int B_  = 128;
constexpr int T_  = 600;
constexpr int H_  = 512;
constexpr int K_  = 10;
constexpr int U_  = 100;
constexpr int CD_ = 80;
constexpr int NG  = 2048;          // 4*H
constexpr int OUT_ = 121;          // 1 + 6*M
// packed A-panel layouts (K padded to multiple of 32):
// A1 = [x(0..2) | w(3..82) | pad(83..95) | h1(96..607)]                      K1 = 608
// A2 = [x(0..2) | h1(3..514) | w(515..594) | pad(595..607) | h2(608..1119)]  K2 = 1120
constexpr int K1 = 608;
constexpr int K2 = 1120;
// LDS row strides (+8 elements => 52 DWORDs mod 64 => conflict-free b128 reads)
constexpr int K1S = K1 + 8;        // 616
constexpr int K2S = K2 + 8;        // 1128

constexpr int NBLK = 128;          // persistent grid: one block per N-tile / per batch row
constexpr int NTHR = 256;          // 8 waves

// ---------------- workspace layout (bytes) ----------------
constexpr size_t OFF_W1T  = 0;
constexpr size_t SZ_W1T   = (size_t)NG * K1 * 2;
constexpr size_t OFF_W2T  = OFF_W1T + SZ_W1T;
constexpr size_t SZ_W2T   = (size_t)NG * K2 * 2;
constexpr size_t OFF_LINT = OFF_W2T + SZ_W2T;                  // [128][512] bf16 (lin_W^T, N padded)
constexpr size_t SZ_LINT  = (size_t)128 * 512 * 2;
constexpr size_t OFF_A1   = OFF_LINT + SZ_LINT;
constexpr size_t SZ_A1    = (size_t)B_ * K1 * 2;
constexpr size_t OFF_A2   = OFF_A1 + SZ_A1;
constexpr size_t SZ_A2    = (size_t)B_ * K2 * 2;
constexpr size_t OFF_G    = OFF_A2 + SZ_A2;                    // gate scratch [128][2048] f32
constexpr size_t SZ_G     = (size_t)B_ * NG * 4;
constexpr size_t OFF_C1   = OFF_G + SZ_G;
constexpr size_t SZ_C     = (size_t)B_ * H_ * 4;
constexpr size_t OFF_C2   = OFF_C1 + SZ_C;
constexpr size_t OFF_KAP  = OFF_C2 + SZ_C;                     // ping-pong [2][128][10] f32
constexpr size_t SZ_KAP   = (size_t)2 * B_ * K_ * 4;
constexpr size_t OFF_BAR  = OFF_KAP + SZ_KAP;                  // grid barrier ticket counter
constexpr size_t SZ_BAR   = 256;
constexpr size_t OFF_YS   = OFF_BAR + SZ_BAR;                  // [B*T][512] bf16 (row = b*T + t)
constexpr size_t SZ_YS    = (size_t)B_ * T_ * H_ * 2;

// ---------------- helpers ----------------
DI float sigm(float v)  { return 1.f / (1.f + __expf(-v)); }
DI float tanh_(float v) { float e = __expf(2.f * v); return 1.f - 2.f / (e + 1.f); }

union FragU { v16bf v; int4 q[2]; };

// One 16x16 output tile: A row-major [M][Kpad] bf16, B pre-transposed [N][Kstride] bf16.
// Fragment layouts per CDNA5 ISA 7.12.2 (16-bit A 16x32, 16-bit B 32x16, wave32):
//   A: lane 0-15 -> M=lane, K-halves 0..7 / 16..23 ; lane 16-31 -> 8..15 / 24..31
//   B: lane 0-15 -> N=lane, K-halves 0..7 / 16..23 ; lane 16-31 -> 8..15 / 24..31
// Both fragments == two contiguous 16B loads per lane (global_load_b128 / ds_load_b128,
// address space inferred after inlining).
template <int KSTEPS, bool PF>
DI v8f wmma_tile(const bf16_t* Ap, const bf16_t* Bp) {
  v8f acc = {0.f, 0.f, 0.f, 0.f, 0.f, 0.f, 0.f, 0.f};
#pragma unroll
  for (int kk = 0; kk < KSTEPS; ++kk) {
    if (PF && kk + 2 < KSTEPS) {
      __builtin_prefetch(Bp + (kk + 2) * 32, 0, 1);   // global_prefetch_b8
    }
    FragU a, b;
    a.q[0] = *(const int4*)(Ap + kk * 32);
    a.q[1] = *(const int4*)(Ap + kk * 32 + 16);
    b.q[0] = *(const int4*)(Bp + kk * 32);
    b.q[1] = *(const int4*)(Bp + kk * 32 + 16);
    acc = __builtin_amdgcn_wmma_f32_16x16x32_bf16(false, a.v, false, b.v,
                                                  (short)0, acc, false, false);
  }
  return acc;
}

// Monotone-ticket grid barrier: counter never resets, barrier k done when cnt >= NBLK*(k+1).
DI void grid_sync(unsigned* bar, unsigned target) {
  __syncthreads();
  if (threadIdx.x == 0) {
    __hip_atomic_fetch_add(bar, 1u, __ATOMIC_RELEASE, __HIP_MEMORY_SCOPE_AGENT);
    while (__hip_atomic_load(bar, __ATOMIC_ACQUIRE, __HIP_MEMORY_SCOPE_AGENT) < target) {
      __builtin_amdgcn_s_sleep(1);
    }
  }
  __syncthreads();
}

// ---------------- kernel 1: pack weights (f32 -> transposed bf16), init state ----------------
__global__ void prep_kernel(char* ws, const float* __restrict__ x,
                            const float* __restrict__ w1k, const float* __restrict__ w1r,
                            const float* __restrict__ w2k, const float* __restrict__ w2r,
                            const float* __restrict__ linW) {
  bf16_t* W1T  = (bf16_t*)(ws + OFF_W1T);
  bf16_t* W2T  = (bf16_t*)(ws + OFF_W2T);
  bf16_t* LINT = (bf16_t*)(ws + OFF_LINT);
  bf16_t* A1   = (bf16_t*)(ws + OFF_A1);
  bf16_t* A2   = (bf16_t*)(ws + OFF_A2);
  float*  C1   = (float*)(ws + OFF_C1);
  float*  C2   = (float*)(ws + OFF_C2);
  float*  KAP  = (float*)(ws + OFF_KAP);
  unsigned* bar = (unsigned*)(ws + OFF_BAR);

  const size_t gid = (size_t)blockIdx.x * blockDim.x + threadIdx.x;
  const size_t gsz = (size_t)gridDim.x * blockDim.x;

  // W1T[n][k]: rows 0..82 <- w1_k, 83..95 zero, 96..607 <- w1_r
  for (size_t i = gid; i < (size_t)NG * K1; i += gsz) {
    int n = (int)(i / K1), k = (int)(i % K1);
    float v = (k < 83) ? w1k[(size_t)k * NG + n]
            : (k >= 96 ? w1r[(size_t)(k - 96) * NG + n] : 0.f);
    W1T[i] = (bf16_t)v;
  }
  // W2T[n][k]: rows 0..594 <- w2_k, 595..607 zero, 608..1119 <- w2_r
  for (size_t i = gid; i < (size_t)NG * K2; i += gsz) {
    int n = (int)(i / K2), k = (int)(i % K2);
    float v = (k < 595) ? w2k[(size_t)k * NG + n]
            : (k >= 608 ? w2r[(size_t)(k - 608) * NG + n] : 0.f);
    W2T[i] = (bf16_t)v;
  }
  // LINT[n][k] = lin_W[k][n], n padded 121 -> 128
  for (size_t i = gid; i < (size_t)128 * 512; i += gsz) {
    int n = (int)(i / 512), k = (int)(i % 512);
    LINT[i] = (bf16_t)(n < OUT_ ? linW[(size_t)k * OUT_ + n] : 0.f);
  }
  // A1: x_0 in slot 0..2, everything else (w, pad, h1) zero
  for (size_t i = gid; i < (size_t)B_ * K1; i += gsz) {
    int b = (int)(i / K1), k = (int)(i % K1);
    A1[i] = (bf16_t)(k < 3 ? x[(size_t)b * T_ * 3 + k] : 0.f);
  }
  for (size_t i = gid; i < (size_t)B_ * K2; i += gsz) A2[i] = (bf16_t)0.f;
  for (size_t i = gid; i < (size_t)B_ * H_; i += gsz) { C1[i] = 0.f; C2[i] = 0.f; }
  for (size_t i = gid; i < (size_t)2 * B_ * K_; i += gsz) KAP[i] = 0.f;
  if (gid == 0) *bar = 0u;
}

// ---------------- kernel 2: persistent recurrence over T steps ----------------
__global__ __launch_bounds__(NTHR, 1)
void rnn_persistent(char* ws, const float* __restrict__ x, const float* __restrict__ c_vec,
                    const float* __restrict__ wwW, const float* __restrict__ wwb,
                    const float* __restrict__ w1b, const float* __restrict__ w2b) {
  const bf16_t* W1T = (const bf16_t*)(ws + OFF_W1T);
  const bf16_t* W2T = (const bf16_t*)(ws + OFF_W2T);
  bf16_t* A1  = (bf16_t*)(ws + OFF_A1);
  bf16_t* A2  = (bf16_t*)(ws + OFF_A2);
  float*  G   = (float*)(ws + OFF_G);
  float*  C1  = (float*)(ws + OFF_C1);
  float*  C2  = (float*)(ws + OFF_C2);
  float*  KAP = (float*)(ws + OFF_KAP);
  bf16_t* YS  = (bf16_t*)(ws + OFF_YS);
  unsigned* bar = (unsigned*)(ws + OFF_BAR);

  const int tid  = threadIdx.x;
  const int bid  = blockIdx.x;        // = N-tile for GEMMs, = batch row for attention
  const int lane = tid & 31;
  const int wv   = tid >> 5;          // 0..7 = M-tile (batch rows wv*16..wv*16+15)
  const int lhi  = lane >> 4;         // 0 = low half-wave, 1 = high
  const int l15  = lane & 15;

  // LDS-resident weight slices for this block's 16 output columns (loaded once,
  // reused for all 600 steps by all 8 waves): 16x608 + 16x1120 bf16, padded rows.
  __shared__ bf16_t w1s[16 * K1S];
  __shared__ bf16_t w2s[16 * K2S];
  __shared__ float kgS[32], eaS[16], ebS[16], knS[16], phiS[112];

  // stage weight slices global -> LDS (16B chunks)
  for (int i = tid; i < 16 * (K1 / 8); i += NTHR) {
    int r = i / (K1 / 8), c = i % (K1 / 8);
    *(int4*)&w1s[r * K1S + c * 8] = *(const int4*)(W1T + (size_t)(bid * 16 + r) * K1 + c * 8);
  }
  for (int i = tid; i < 16 * (K2 / 8); i += NTHR) {
    int r = i / (K2 / 8), c = i % (K2 / 8);
    *(int4*)&w2s[r * K2S + c * 8] = *(const int4*)(W2T + (size_t)(bid * 16 + r) * K2 + c * 8);
  }
  __syncthreads();

  unsigned target = 0;

  for (int t = 0; t < T_; ++t) {
    // -------- phase 1: LSTM1 gate GEMM  G = A1 @ W1  (128x2048, K=608) --------
    {
      const bf16_t* Ap = A1 + (size_t)(wv * 16 + l15) * K1 + lhi * 8;
      const bf16_t* Bp = w1s + l15 * K1S + lhi * 8;       // LDS (ds_load_b128)
      v8f acc = wmma_tile<K1 / 32, false>(Ap, Bp);
      float* Gp = G + (size_t)(wv * 16 + lhi * 8) * NG + bid * 16 + l15;
#pragma unroll
      for (int r = 0; r < 8; ++r) Gp[(size_t)r * NG] = acc[r];
    }
    target += NBLK; grid_sync(bar, target);

    // -------- phase 2: LSTM1 activations -> c1, h1 (h1 into A1 & A2 panels) --------
    for (int e = bid * NTHR + tid; e < B_ * H_; e += NBLK * NTHR) {
      int b = e >> 9, j = e & 511;
      const float* g = G + (size_t)b * NG;
      float zi = g[j]          + w1b[j];
      float zf = g[j + 512]    + w1b[j + 512];
      float zg = g[j + 1024]   + w1b[j + 1024];
      float zo = g[j + 1536]   + w1b[j + 1536];
      float c  = sigm(zf) * C1[e] + sigm(zi) * tanh_(zg);
      C1[e] = c;
      bf16_t h = (bf16_t)(sigm(zo) * tanh_(c));
      A1[(size_t)b * K1 + 96 + j] = h;
      A2[(size_t)b * K2 + 3 + j]  = h;
    }
    target += NBLK; grid_sync(bar, target);

    // -------- phase 3: attention window (block bid owns batch row b=bid) --------
    {
      const int b = bid;
      const bf16_t* h1 = A2 + (size_t)b * K2 + 3;
      // kg = h1 @ ww_W + ww_b : wave wv owns outputs j = wv*4 .. wv*4+3 (30 valid),
      // lanes stride K, 5-step shfl_xor butterfly reduction.
      {
        const int j = wv * 4;
        const int j0 = (j + 0 < 30) ? j + 0 : 29;
        const int j1 = (j + 1 < 30) ? j + 1 : 29;
        const int j2 = (j + 2 < 30) ? j + 2 : 29;
        const int j3 = (j + 3 < 30) ? j + 3 : 29;
        float p0 = 0.f, p1 = 0.f, p2 = 0.f, p3 = 0.f;
        for (int k2 = lane; k2 < H_; k2 += 32) {
          float hv = (float)h1[k2];
          const float* wr = wwW + (size_t)k2 * 30;
          p0 += hv * wr[j0];
          p1 += hv * wr[j1];
          p2 += hv * wr[j2];
          p3 += hv * wr[j3];
        }
#pragma unroll
        for (int m = 16; m >= 1; m >>= 1) {
          p0 += __shfl_xor(p0, m, 32);
          p1 += __shfl_xor(p1, m, 32);
          p2 += __shfl_xor(p2, m, 32);
          p3 += __shfl_xor(p3, m, 32);
        }
        if (lane == 0) {
          if (j + 0 < 30) kgS[j + 0] = p0 + wwb[j + 0];
          if (j + 1 < 30) kgS[j + 1] = p1 + wwb[j + 1];
          if (j + 2 < 30) kgS[j + 2] = p2 + wwb[j + 2];
          if (j + 3 < 30) kgS[j + 3] = p3 + wwb[j + 3];
        }
      }
      __syncthreads();
      if (tid < K_) {                               // kappa += exp(k_hat)
        float kold = KAP[((size_t)(t & 1) * B_ + b) * K_ + tid];
        float kn   = kold + __expf(kgS[20 + tid]);
        KAP[((size_t)((t + 1) & 1) * B_ + b) * K_ + tid] = kn;
        knS[tid] = kn;
        eaS[tid] = __expf(kgS[tid]);
        ebS[tid] = __expf(kgS[10 + tid]);
      }
      __syncthreads();
      if (tid < U_) {                               // phi[u]
        float u = (float)tid, s = 0.f;
        for (int k2 = 0; k2 < K_; ++k2) {
          float d = knS[k2] - u;
          s += eaS[k2] * __expf(-ebS[k2] * d * d);
        }
        phiS[tid] = s;
      }
      __syncthreads();
      if (tid < CD_) {                              // w = phi @ c_vec[b]
        float s = 0.f;
        const float* cv = c_vec + (size_t)b * U_ * CD_ + tid;
        for (int u2 = 0; u2 < U_; ++u2) s += phiS[u2] * cv[(size_t)u2 * CD_];
        bf16_t wbf = (bf16_t)s;
        A1[(size_t)b * K1 + 3 + tid]   = wbf;       // for LSTM1 at t+1
        A2[(size_t)b * K2 + 515 + tid] = wbf;       // for LSTM2 at t
      }
      if (tid >= 104 && tid < 107) {                // x slots: A2 <- x_t ; A1 <- x_{t+1}
        int d = tid - 104;
        A2[(size_t)b * K2 + d] = (bf16_t)x[((size_t)b * T_ + t) * 3 + d];
        float xn = (t + 1 < T_) ? x[((size_t)b * T_ + t + 1) * 3 + d] : 0.f;
        A1[(size_t)b * K1 + d] = (bf16_t)xn;
      }
    }
    target += NBLK; grid_sync(bar, target);

    // -------- phase 4: LSTM2 gate GEMM  G = A2 @ W2  (128x2048, K=1120) --------
    {
      const bf16_t* Ap = A2 + (size_t)(wv * 16 + l15) * K2 + lhi * 8;
      const bf16_t* Bp = w2s + l15 * K2S + lhi * 8;       // LDS (ds_load_b128)
      v8f acc = wmma_tile<K2 / 32, false>(Ap, Bp);
      float* Gp = G + (size_t)(wv * 16 + lhi * 8) * NG + bid * 16 + l15;
#pragma unroll
      for (int r = 0; r < 8; ++r) Gp[(size_t)r * NG] = acc[r];
    }
    target += NBLK; grid_sync(bar, target);

    // -------- phase 5: LSTM2 activations -> c2, h2; emit h2 to ys --------
    for (int e = bid * NTHR + tid; e < B_ * H_; e += NBLK * NTHR) {
      int b = e >> 9, j = e & 511;
      const float* g = G + (size_t)b * NG;
      float zi = g[j]          + w2b[j];
      float zf = g[j + 512]    + w2b[j + 512];
      float zg = g[j + 1024]   + w2b[j + 1024];
      float zo = g[j + 1536]   + w2b[j + 1536];
      float c  = sigm(zf) * C2[e] + sigm(zi) * tanh_(zg);
      C2[e] = c;
      bf16_t h = (bf16_t)(sigm(zo) * tanh_(c));
      A2[(size_t)b * K2 + 608 + j] = h;                       // recurrent input at t+1
      YS[((size_t)b * T_ + t) * H_ + j] = h;                  // flat row = b*T + t
    }
    target += NBLK; grid_sync(bar, target);
  }
}

// ---------------- kernel 3: head GEMM  out[76800][121] = ys @ lin_W + lin_b ----------------
__global__ __launch_bounds__(256)
void head_gemm(const char* ws, const float* __restrict__ linb, float* __restrict__ out) {
  const bf16_t* YS   = (const bf16_t*)(ws + OFF_YS);
  const bf16_t* LINT = (const bf16_t*)(ws + OFF_LINT);
  const int lane = threadIdx.x & 31;
  const int wv   = threadIdx.x >> 5;  // N-tile 0..7 (cols wv*16..wv*16+15, padded to 128)
  const int lhi  = lane >> 4;
  const int l15  = lane & 15;
  const int mt   = blockIdx.x;        // M-tile 0..4799

  const bf16_t* Ap = YS + (size_t)(mt * 16 + l15) * H_ + lhi * 8;
  const bf16_t* Bp = LINT + (size_t)(wv * 16 + l15) * H_ + lhi * 8;
  v8f acc = wmma_tile<H_ / 32, true>(Ap, Bp);

  int col = wv * 16 + l15;
  if (col < OUT_) {
    float bias = linb[col];
    int row0 = mt * 16 + lhi * 8;
#pragma unroll
    for (int r = 0; r < 8; ++r) out[(size_t)(row0 + r) * OUT_ + col] = acc[r] + bias;
  }
}

// ---------------- entry ----------------
extern "C" void kernel_launch(void* const* d_in, const int* in_sizes, int n_in,
                              void* d_out, int out_size, void* d_ws, size_t ws_size,
                              hipStream_t stream) {
  const float* x     = (const float*)d_in[0];
  const float* c_vec = (const float*)d_in[1];
  const float* w1k   = (const float*)d_in[2];
  const float* w1r   = (const float*)d_in[3];
  const float* w1b   = (const float*)d_in[4];
  const float* w2k   = (const float*)d_in[5];
  const float* w2r   = (const float*)d_in[6];
  const float* w2b   = (const float*)d_in[7];
  const float* wwW   = (const float*)d_in[8];
  const float* wwb   = (const float*)d_in[9];
  const float* linW  = (const float*)d_in[10];
  const float* linb  = (const float*)d_in[11];
  char* ws = (char*)d_ws;

  prep_kernel<<<256, 256, 0, stream>>>(ws, x, w1k, w1r, w2k, w2r, linW);
  rnn_persistent<<<NBLK, NTHR, 0, stream>>>(ws, x, c_vec, wwW, wwb, w1b, w2b);
  head_gemm<<<B_ * T_ / 16, 256, 0, stream>>>(ws, linb, (float*)d_out);
}
